// TreeLSTMClassifier_26491358282208
// MI455X (gfx1250) — matrix operand order repaired and checked by reading, not compiled, not run on hardware
//
#include <hip/hip_runtime.h>

// ---------------- CDNA5 WMMA types ----------------
typedef __attribute__((ext_vector_type(16))) __bf16 v16bf;
typedef __attribute__((ext_vector_type(8)))  float  v8f;

union LdFrag { uint4 q[2]; v16bf v; };  // 32B <-> 16 bf16

// ---- CDNA5 hardware tanh (TRANS32 op, no clang builtin) ----
// TRANS result needs 1 delay before use (ISA 7.4): embed s_delay_alu.
__device__ __forceinline__ float tanh_hw(float x) {
  float r;
  asm("v_tanh_f32 %0, %1\n\t"
      "s_delay_alu instid0(TRANS32_DEP_1)"
      : "=v"(r) : "v"(x));
  return r;
}
// sigmoid(x) = 0.5*tanh(x/2) + 0.5
__device__ __forceinline__ float sigm_hw(float x) {
  return __builtin_fmaf(0.5f, tanh_hw(0.5f * x), 0.5f);
}

// ---- A fragment (16x32 bf16), wave32 layout:
// lane L holds row m=L&15; VGPRs 0..3: K = half*8..half*8+7,
// VGPRs 4..7: K = 16+half*8..16+half*8+7, where half = L>>4.
// kb = kstep + half*8. Elements [kb..kb+7] and [kb+16..kb+23].

// From a bf16 row (tree levels: h stored in bf16) -> two raw b128 loads.
__device__ __forceinline__ v16bf a_frag_bf16(const __bf16* __restrict__ row, int kb) {
  LdFrag t;
  t.q[0] = *(const uint4*)(row + kb);
  t.q[1] = *(const uint4*)(row + kb + 16);
  return t.v;
}

// From an f32 row (leaf: embedding) with hardware packed cvt.
__device__ __forceinline__ v16bf a_frag_f32(const float* __restrict__ row, int kb) {
  float4 s0 = *(const float4*)(row + kb);
  float4 s1 = *(const float4*)(row + kb + 4);
  float4 s2 = *(const float4*)(row + kb + 16);
  float4 s3 = *(const float4*)(row + kb + 20);
  v16bf a;
  a[0]=(__bf16)s0.x;  a[1]=(__bf16)s0.y;  a[2]=(__bf16)s0.z;  a[3]=(__bf16)s0.w;
  a[4]=(__bf16)s1.x;  a[5]=(__bf16)s1.y;  a[6]=(__bf16)s1.z;  a[7]=(__bf16)s1.w;
  a[8]=(__bf16)s2.x;  a[9]=(__bf16)s2.y;  a[10]=(__bf16)s2.z; a[11]=(__bf16)s2.w;
  a[12]=(__bf16)s3.x; a[13]=(__bf16)s3.y; a[14]=(__bf16)s3.z; a[15]=(__bf16)s3.w;
  return a;
}

// Guarded variant for the leaf K tail (E=300, loop padded to 320).
__device__ __forceinline__ v16bf a_frag_f32_guard(const float* __restrict__ row,
                                                  int kb, int limit) {
  v16bf a;
#pragma unroll
  for (int i = 0; i < 8; ++i) {
    int e0 = kb + i, e1 = kb + 16 + i;
    a[i]     = (__bf16)(e0 < limit ? row[e0] : 0.0f);
    a[8 + i] = (__bf16)(e1 < limit ? row[e1] : 0.0f);
  }
  return a;
}

// ---- B fragment (32x16 bf16): lane L holds col n=L&15; same K layout as A:
// VGPRs 0..3: K = half*8..half*8+7, VGPRs 4..7: K = 16+half*8..16+half*8+7.
// => bf16 [kb..kb+7] and [kb+16..kb+23] from the U row at kb = kstep + (L>>4)*8.
__device__ __forceinline__ v16bf b_frag_bf16(const __bf16* __restrict__ row, int kb) {
  LdFrag t;
  t.q[0] = *(const uint4*)(row + kb);
  t.q[1] = *(const uint4*)(row + kb + 16);
  return t.v;
}

#define WMMA_BF16(A, Bm, C) \
  __builtin_amdgcn_wmma_f32_16x16x32_bf16(false, (A), false, (Bm), (short)0, (C), false, false)

// ---------------- setup: convert weights to bf16, build bias ----------------
// Wbf: [4][256][320] (E padded 300->320, zero tail); ULbf/URbf: [5][256][256]
// bias[g][k] = bW[GMAP[g]][k] + bUL[g][k] + bUR[g][k]
__global__ void tl_setup_kernel(const float* __restrict__ W,  const float* __restrict__ bW,
                                const float* __restrict__ UL, const float* __restrict__ bUL,
                                const float* __restrict__ UR, const float* __restrict__ bUR,
                                __bf16* __restrict__ Wbf,
                                __bf16* __restrict__ ULbf,
                                __bf16* __restrict__ URbf,
                                float* __restrict__ bias) {
  const int NW = 4 * 256 * 320;
  const int NU = 5 * 256 * 256;
  const int NB = 5 * 256;
  const int total = NW + 2 * NU + NB;
  for (int t = blockIdx.x * blockDim.x + threadIdx.x; t < total;
       t += gridDim.x * blockDim.x) {
    if (t < NW) {
      int e = t % 320;
      int rk = t / 320;  // wrow*256 + k
      Wbf[t] = (__bf16)(e < 300 ? W[(size_t)rk * 300 + e] : 0.0f);
    } else if (t < NW + NU) {
      int u = t - NW;
      ULbf[u] = (__bf16)UL[u];
    } else if (t < NW + 2 * NU) {
      int u = t - NW - NU;
      URbf[u] = (__bf16)UR[u];
    } else {
      int u = t - NW - 2 * NU;
      int g = u >> 8, k = u & 255;
      const int gm[5] = {0, 1, 1, 2, 3};
      bias[u] = bW[gm[g] * 256 + k] + bUL[u] + bUR[u];
    }
  }
}

// ---------------- leaf: gather emb + GEMM (i,o,u gates) + activation ----------------
// One wave: 16-row x 16-col tile for gates {i,o,u} (W rows {0,2,3}).
// h written in bf16 (feeds next level's A), c in f32.
__global__ __launch_bounds__(256) void tl_leaf_kernel(
    const int* __restrict__ tokens, const float* __restrict__ emb,
    const __bf16* __restrict__ Wbf, const float* __restrict__ bias,
    __bf16* __restrict__ hout, float* __restrict__ cout) {
  const int H = 256, E = 300, EP = 320;
  int wave = threadIdx.x >> 5, lane = threadIdx.x & 31;
  int gw = blockIdx.x * 8 + wave;
  int ctile = gw & 15, mtile = gw >> 4;

  int hlf = lane >> 4, n = lane & 15;
  int col = ctile * 16 + n;

  int tok = tokens[mtile * 16 + (lane & 15)];
  const float* arow = emb + (size_t)tok * E;

  const int wrow[3] = {0, 2, 3};
  const __bf16* bp[3];
#pragma unroll
  for (int g = 0; g < 3; ++g)
    bp[g] = Wbf + (size_t)(wrow[g] * H + col) * EP;

  const v8f vz = {0.f, 0.f, 0.f, 0.f, 0.f, 0.f, 0.f, 0.f};
  v8f acc[3] = {vz, vz, vz};

  int akb = hlf * 8, bkb = hlf * 8;
  for (int k = 0; k < EP; k += 32) {
    v16bf a = (k + 32 <= E) ? a_frag_f32(arow, k + akb)
                            : a_frag_f32_guard(arow, k + akb, E);
#pragma unroll
    for (int g = 0; g < 3; ++g) {
      v16bf bm = b_frag_bf16(bp[g], k + bkb);
      acc[g] = WMMA_BF16(a, bm, acc[g]);
    }
  }

  float b0 = bias[0 * H + col], b3 = bias[3 * H + col], b4 = bias[4 * H + col];
#pragma unroll
  for (int r = 0; r < 8; ++r) {
    int mloc = mtile * 16 + r + 8 * hlf;  // C layout: VGPR r -> row r (+8 for lanes 16-31)
    float iv = acc[0][r] + b0;
    float ov = acc[1][r] + b3;
    float uv = acc[2][r] + b4;
    float cn = sigm_hw(iv) * tanh_hw(uv);
    float hn = sigm_hw(ov) * tanh_hw(cn);
    hout[(size_t)mloc * H + col] = (__bf16)hn;
    cout[(size_t)mloc * H + col] = cn;
  }
}

// ---------------- tree level: pre = hl*UL^T + hr*UR^T + bias; LSTM cell ----------------
// One wave: 16 output rows x 16 cols, all 5 gates (40 accum VGPRs).
// h in/out bf16; c in/out f32. On the last level also writes f32 h (root).
__global__ __launch_bounds__(256) void tl_level_kernel(
    const __bf16* __restrict__ hin, const float* __restrict__ cin,
    __bf16* __restrict__ hout, float* __restrict__ cout,
    const __bf16* __restrict__ ULbf, const __bf16* __restrict__ URbf,
    const float* __restrict__ bias, int n_out, int shift,
    float* __restrict__ hrootf) {
  const int H = 256;
  int wave = threadIdx.x >> 5, lane = threadIdx.x & 31;
  int gw = blockIdx.x * 8 + wave;
  int ctile = gw & 15, mtile = gw >> 4;

  int hlf = lane >> 4, n = lane & 15;
  int col = ctile * 16 + n;

  // A rows for this lane (children of output row ma)
  int ma = mtile * 16 + (lane & 15);
  int bidx = ma >> shift;
  int j = ma & (n_out - 1);
  int rowL = (bidx << (shift + 1)) + 2 * j;
  const __bf16* arowL = hin + (size_t)rowL * H;
  const __bf16* arowR = arowL + H;

  const __bf16* bl[5];
  const __bf16* br[5];
#pragma unroll
  for (int g = 0; g < 5; ++g) {
    bl[g] = ULbf + (size_t)(g * H + col) * H;
    br[g] = URbf + (size_t)(g * H + col) * H;
  }

  const v8f vz = {0.f, 0.f, 0.f, 0.f, 0.f, 0.f, 0.f, 0.f};
  v8f acc[5] = {vz, vz, vz, vz, vz};

  int akb = hlf * 8, bkb = hlf * 8;
  for (int k = 0; k < H; k += 32) {
    v16bf a = a_frag_bf16(arowL, k + akb);
#pragma unroll
    for (int g = 0; g < 5; ++g) {
      v16bf bm = b_frag_bf16(bl[g], k + bkb);
      acc[g] = WMMA_BF16(a, bm, acc[g]);
    }
    a = a_frag_bf16(arowR, k + akb);
#pragma unroll
    for (int g = 0; g < 5; ++g) {
      v16bf bm = b_frag_bf16(br[g], k + bkb);
      acc[g] = WMMA_BF16(a, bm, acc[g]);
    }
  }

  float b0 = bias[0 * H + col], b1 = bias[1 * H + col], b2 = bias[2 * H + col];
  float b3 = bias[3 * H + col], b4 = bias[4 * H + col];
#pragma unroll
  for (int r = 0; r < 8; ++r) {
    int mloc = mtile * 16 + r + 8 * hlf;
    int bb = mloc >> shift;
    int jj = mloc & (n_out - 1);
    int rL = (bb << (shift + 1)) + 2 * jj;
    float cl = cin[(size_t)rL * H + col];
    float cr = cin[(size_t)(rL + 1) * H + col];
    float iv = acc[0][r] + b0;
    float fl = acc[1][r] + b1;
    float fr = acc[2][r] + b2;
    float ov = acc[3][r] + b3;
    float uv = acc[4][r] + b4;
    float cn = sigm_hw(iv) * tanh_hw(uv) + sigm_hw(fl) * cl + sigm_hw(fr) * cr;
    float hn = sigm_hw(ov) * tanh_hw(cn);
    hout[(size_t)mloc * H + col] = (__bf16)hn;
    cout[(size_t)mloc * H + col] = cn;
    if (hrootf) hrootf[(size_t)mloc * H + col] = hn;  // uniform branch, last level only
  }
}

// ---------------- final: root h (f32), c (f32) -> d_out (h first, then c) -------------
__global__ void tl_out_kernel(const float* __restrict__ h, const float* __restrict__ c,
                              float* __restrict__ out) {
  int i = blockIdx.x * blockDim.x + threadIdx.x;
  if (i < 64 * 256) {
    out[i] = h[i];
    out[64 * 256 + i] = c[i];
  }
}

extern "C" void kernel_launch(void* const* d_in, const int* in_sizes, int n_in,
                              void* d_out, int out_size, void* d_ws, size_t ws_size,
                              hipStream_t stream) {
  (void)in_sizes; (void)n_in; (void)out_size; (void)ws_size;
  const int*   tokens = (const int*)  d_in[0];
  const float* emb    = (const float*)d_in[1];
  const float* W      = (const float*)d_in[2];
  const float* bW     = (const float*)d_in[3];
  const float* UL     = (const float*)d_in[4];
  const float* bUL    = (const float*)d_in[5];
  const float* UR     = (const float*)d_in[6];
  const float* bUR    = (const float*)d_in[7];
  float* out = (float*)d_out;

  // Workspace carve-out (256B aligned)
  char* base = (char*)d_ws;
  size_t off = 0;
  auto take = [&](size_t bytes) -> char* {
    char* r = base + off;
    off = (off + bytes + 255) & ~(size_t)255;
    return r;
  };
  __bf16* Wbf  = (__bf16*)take((size_t)4 * 256 * 320 * 2);
  __bf16* ULbf = (__bf16*)take((size_t)5 * 256 * 256 * 2);
  __bf16* URbf = (__bf16*)take((size_t)5 * 256 * 256 * 2);
  float*  bias = (float*)take((size_t)5 * 256 * 4);
  __bf16* h0 = (__bf16*)take((size_t)65536 * 256 * 2);  // bf16 h ping
  float*  c0 = (float*) take((size_t)65536 * 256 * 4);  // f32 c ping
  __bf16* h1 = (__bf16*)take((size_t)32768 * 256 * 2);  // bf16 h pong
  float*  c1 = (float*) take((size_t)32768 * 256 * 4);  // f32 c pong
  float*  hroot = (float*)take((size_t)64 * 256 * 4);   // f32 root h

  tl_setup_kernel<<<512, 256, 0, stream>>>(W, bW, UL, bUL, UR, bUR,
                                           Wbf, ULbf, URbf, bias);

  // Leaf: M=65536 rows, 16 col tiles -> 65536 waves -> 8192 blocks of 8 waves
  tl_leaf_kernel<<<8192, 256, 0, stream>>>(tokens, emb, Wbf, bias, h0, c0);

  const __bf16* hin = h0;
  const float*  cin = c0;
  __bf16* ha = h1; float* ca = c1;  // current write target
  __bf16* hb = h0; float* cb = c0;  // next write target
  int n_out = 512, shift = 9;
  for (int lvl = 0; lvl < 10; ++lvl) {
    int M = 64 * n_out;
    int blocks = (M / 16) * 16 / 8;  // waves / 8, always integral (M >= 64)
    float* hr = (lvl == 9) ? hroot : nullptr;
    tl_level_kernel<<<blocks, 256, 0, stream>>>(hin, cin, ha, ca,
                                                ULbf, URbf, bias, n_out, shift, hr);
    hin = ha; cin = ca;
    __bf16* th = ha; ha = hb; hb = th;
    float*  tc = ca; ca = cb; cb = tc;
    n_out >>= 1;
    shift -= 1;
  }

  tl_out_kernel<<<64, 256, 0, stream>>>(hroot, cin, out);
}
